// RNNDecoder_58360015618646
// MI455X (gfx1250) — matrix operand order repaired and checked by reading, not compiled, not run on hardware
//
#include <hip/hip_runtime.h>
#include <hip/hip_bf16.h>

typedef unsigned short u16;
typedef __attribute__((ext_vector_type(16))) __bf16 v16bf;
typedef __attribute__((ext_vector_type(8)))  float  v8f;

#define Bsz 128
#define Pn  512
#define Hn  512
#define Sn  512
#define Dn  15
#define En  256
#define TH3 1536   // 3*H
#define TARP 1023  // P + S - 1

// ---------- helpers ----------
__device__ __forceinline__ u16 f2b(float x) {
    unsigned u = __float_as_uint(x);
    unsigned r = (u + 0x7FFFu + ((u >> 16) & 1u)) >> 16;
    return (u16)r;
}
// WMMA bf16 A-matrix 16x32 lane/slot -> K mapping (ISA 7.12.2)
__device__ __forceinline__ int a_k_of(int lane, int s) {
    return s + ((s >= 8) ? 8 : 0) + ((lane >= 16) ? 8 : 0);
}
// WMMA bf16 B-matrix 32x16: K = a_k_of(lane, s) (same as A), N = lane & 15
__device__ __forceinline__ v8f wmma_bf16(v16bf a, v16bf b, v8f c) {
    return __builtin_amdgcn_wmma_f32_16x16x32_bf16(false, a, false, b, (short)0, c, false, false);
}
__device__ __forceinline__ float sigmoidf_fast(float x) {
    return 1.0f / (1.0f + __expf(-x));
}
__device__ __forceinline__ float tanhf_fast(float x) {
    float t = __expf(-2.0f * fabsf(x));
    float th = (1.0f - t) / (1.0f + t);
    return copysignf(th, x);
}
// Opaque zero: regenerated every loop iteration so weight-fragment addresses are
// not loop-invariant -> compiler cannot hoist (and then spill) the L2 streams.
__device__ __forceinline__ size_t opaque_zero() {
    unsigned z;
    asm volatile("s_mov_b32 %0, 0" : "=s"(z));
    return (size_t)z;
}

// ---------- setup: fuse Wemb@gru_k ----------
__global__ void k_wfused(const float* __restrict__ Wemb, const float* __restrict__ gru_k,
                         float* __restrict__ Wf /*15x1536*/) {
    int i = blockIdx.x * blockDim.x + threadIdx.x;
    if (i >= Dn * TH3) return;
    int d = i / TH3, n = i % TH3;
    float acc = 0.f;
    for (int e = 0; e < En; ++e) acc += Wemb[d * En + e] * gru_k[(size_t)e * TH3 + n];
    Wf[i] = acc;
}
__global__ void k_biasx(const float* __restrict__ bemb, const float* __restrict__ gru_k,
                        const float* __restrict__ gru_b, float* __restrict__ bias_x) {
    int n = blockIdx.x * blockDim.x + threadIdx.x;
    if (n >= TH3) return;
    float a = gru_b[n];
    for (int e = 0; e < En; ++e) a += bemb[e] * gru_k[(size_t)e * TH3 + n];
    bias_x[n] = a;
}

// ---------- setup: pack row-major [K][N] weight into WMMA B fragments ----------
// frag layout: frag[(nt*ktiles + kt)*512 + lane*16 + s], bf16
__global__ void k_pack_b(const float* __restrict__ W, int K, int N, int ld,
                         int ktiles, int ntiles, u16* __restrict__ out) {
    int idx = blockIdx.x * blockDim.x + threadIdx.x;
    int total = ntiles * ktiles * 512;
    if (idx >= total) return;
    int frag = idx >> 9;
    int e = idx & 511;
    int lane = e >> 4, s = e & 15;
    int kt = frag % ktiles, nt = frag / ktiles;
    int k = kt * 32 + ((lane < 16) ? 0 : 8) + s + ((s >= 8) ? 8 : 0);
    int n = nt * 16 + (lane & 15);
    float v = (k < K && n < N) ? W[(size_t)k * ld + n] : 0.f;
    out[idx] = f2b(v);
}

// ---------- attention (one block per batch row) ----------
__global__ __launch_bounds__(256) void k_attention(const float* __restrict__ hs,
                                                   const float* __restrict__ masks,
                                                   const float* __restrict__ Wq,
                                                   const float* __restrict__ bq,
                                                   float* __restrict__ attn) {
    __shared__ float sc[Pn];
    __shared__ float red[256];
    int b = blockIdx.x, tid = threadIdx.x;
    const float* hb = hs + (size_t)b * Pn * Hn;
    for (int p = tid; p < Pn; p += 256) {
        const float* row = hb + (size_t)p * Hn;
        float s = 0.f;
        for (int h = 0; h < Hn; ++h) s += row[h] * Wq[h];
        s += bq[0];
        s += masks[(size_t)b * Pn + p] * -1000000000.0f;
        sc[p] = s;
    }
    __syncthreads();
    float m = -3.4e38f;
    for (int p = tid; p < Pn; p += 256) m = fmaxf(m, sc[p]);
    red[tid] = m; __syncthreads();
    for (int o = 128; o > 0; o >>= 1) { if (tid < o) red[tid] = fmaxf(red[tid], red[tid + o]); __syncthreads(); }
    m = red[0]; __syncthreads();
    float psum = 0.f;
    for (int p = tid; p < Pn; p += 256) { float e = __expf(sc[p] - m); sc[p] = e; psum += e; }
    red[tid] = psum; __syncthreads();
    for (int o = 128; o > 0; o >>= 1) { if (tid < o) red[tid] += red[tid + o]; __syncthreads(); }
    float inv = 1.0f / red[0]; __syncthreads();
    for (int h = tid; h < Hn; h += 256) {
        float acc = 0.f;
        for (int p = 0; p < Pn; ++p) acc += sc[p] * hb[(size_t)p * Hn + h];
        attn[(size_t)b * Hn + h] = acc * inv;
    }
}

// ---------- tar[:, :P, :] = hidden_states (strided copy, float4) ----------
__global__ void k_copy_hs(const float4* __restrict__ hs, float4* __restrict__ tar) {
    size_t i = (size_t)blockIdx.x * blockDim.x + threadIdx.x;
    size_t total = (size_t)Bsz * Pn * Hn / 4;
    if (i >= total) return;
    size_t per_b = (size_t)Pn * Hn / 4;       // 65536
    size_t b = i / per_b, r = i % per_b;
    tar[b * ((size_t)TARP * Hn / 4) + r] = hs[i];
}

// ---------- persistent GRU scan: 8 WGs x 16 batch rows, 32 waves (1 H-tile/wave) ----------
__global__ __launch_bounds__(1024) void k_scan(const float* __restrict__ hs,
                                               const float* __restrict__ init_point,
                                               const float* __restrict__ attn,
                                               const u16* __restrict__ wfFrag,   // 96 frags (K=15 padded)
                                               const u16* __restrict__ rkFrag,   // 96 x 16 frags
                                               const u16* __restrict__ wlFrag,   // 32 frags
                                               const float* __restrict__ bias_x,
                                               const float* __restrict__ gru_b,
                                               const float* __restrict__ blin,
                                               float* __restrict__ seq,
                                               float* __restrict__ tar) {
    __shared__ __align__(32) u16 aState[16][32][16]; // 16 KB, state A-fragments (K=512)
    __shared__ __align__(32) u16 aAttn[16][32][16];  // 16 KB, attn A-fragments (constant)
    __shared__ __align__(32) u16 aOut[32][16];       // 1 KB, out A-fragment (K=15 padded to 32)

    const int tid = threadIdx.x, lane = tid & 31, wave = tid >> 5;  // wave = H-tile (0..31)
    const int b0 = blockIdx.x * 16;

    // ---- prologue: build attn + initial state fragments ----
    for (int i = tid; i < 16 * 512; i += 1024) {
        int kt = i >> 9, e = i & 511;
        int l = e >> 4, s = e & 15;
        int row = l & 15;
        int col = kt * 32 + a_k_of(l, s);
        aAttn[kt][l][s]  = f2b(attn[(size_t)(b0 + row) * Hn + col]);
        aState[kt][l][s] = f2b(hs[((size_t)(b0 + row) * Pn + (Pn - 1)) * Hn + col]);
    }
    if (tid < 512) {
        int l = tid >> 4, s = tid & 15;
        int row = l & 15, kk = a_k_of(l, s);
        float v = (kk < Dn) ? init_point[(b0 + row) * Dn + kk] : 0.f;
        aOut[l][s] = f2b(v);
    }
    for (int i = tid; i < 16 * Dn; i += 1024) {  // seq[:,0,:] = init_point
        int row = i / Dn, d = i % Dn;
        seq[((size_t)(b0 + row) * Sn) * Dn + d] = init_point[(b0 + row) * Dn + d];
    }

    // wave owns H-tile ht == wave; fp32 state lives in C-fragment registers
    const int ht = wave;
    const int col = ht * 16 + (lane & 15);
    v8f stC;
    for (int v = 0; v < 8; ++v) {
        int row = v + ((lane >= 16) ? 8 : 0);
        stC[v] = hs[((size_t)(b0 + row) * Pn + (Pn - 1)) * Hn + col];
    }
    const float bZ  = bias_x[col]          + gru_b[TH3 + col];
    const float bR  = bias_x[col + Hn]     + gru_b[TH3 + col + Hn];
    const float bXH = bias_x[col + 2 * Hn];
    const float bHH = gru_b[TH3 + col + 2 * Hn];
    float bO = 0.f;
    { int c = lane & 15; if (c < Dn) bO = blin[c]; }

    // per-wave fragment base offsets (elements)
    const size_t wfZo = ((size_t)(ht)      ) * 512 + lane * 16;
    const size_t wfRo = ((size_t)(ht + 32) ) * 512 + lane * 16;
    const size_t wfHo = ((size_t)(ht + 64) ) * 512 + lane * 16;
    const size_t rkZo = ((size_t)(ht)      * 16) * 512 + lane * 16;
    const size_t rkRo = ((size_t)(ht + 32) * 16) * 512 + lane * 16;
    const size_t rkHo = ((size_t)(ht + 64) * 16) * 512 + lane * 16;

    // precomputed LDS write slot for the state update (fixed per lane/wave)
    const int wkt = col >> 5;
    const int wkk = col & 31;
    const int wl  = 16 * ((wkk >> 3) & 1);
    const int ws_ = (wkk < 16) ? (wkk & 7) : (8 + (wkk & 7));

    __syncthreads();

    // ---- 511 sequential steps ----
    for (int t = 0; t < Sn - 1; ++t) {
        // Opaque per-iteration zero keeps all weight loads inside the loop
        // (streamed from L2 every step) instead of hoisted + spilled to scratch.
        const size_t zz = opaque_zero();

        v8f accZ, accR, accXH, accHH;
        for (int v = 0; v < 8; ++v) {
            accZ[v] = bZ; accR[v] = bR; accXH[v] = bXH; accHH[v] = bHH;
        }

        // x-GEMM: out(16x15) @ Wfused(15x1536), single K-tile
        {
            v16bf aO = *(const v16bf*)&aOut[lane][0];
            accZ  = wmma_bf16(aO, *(const v16bf*)(wfFrag + wfZo + zz), accZ);
            accR  = wmma_bf16(aO, *(const v16bf*)(wfFrag + wfRo + zz), accR);
            accXH = wmma_bf16(aO, *(const v16bf*)(wfFrag + wfHo + zz), accXH);
        }
        // h-GEMM: state(16x512) @ gru_rk(512x1536), 16 K-tiles (L2-resident weights),
        // manually double-buffered so each k-tile's B loads hide under the previous WMMAs
        {
            const u16* pz = rkFrag + rkZo + zz;
            const u16* pr = rkFrag + rkRo + zz;
            const u16* ph = rkFrag + rkHo + zz;
            v16bf bz = *(const v16bf*)pz;
            v16bf br = *(const v16bf*)pr;
            v16bf bh = *(const v16bf*)ph;
            for (int kt = 0; kt < 16; ++kt) {
                v16bf aS = *(const v16bf*)&aState[kt][lane][0];
                v16bf bz2, br2, bh2;
                if (kt < 15) {
                    __builtin_prefetch(pz + 1024, 0, 3);
                    bz2 = *(const v16bf*)(pz + 512);
                    br2 = *(const v16bf*)(pr + 512);
                    bh2 = *(const v16bf*)(ph + 512);
                }
                accZ  = wmma_bf16(aS, bz, accZ);
                accR  = wmma_bf16(aS, br, accR);
                accHH = wmma_bf16(aS, bh, accHH);
                bz = bz2; br = br2; bh = bh2;
                pz += 512; pr += 512; ph += 512;
            }
        }
        __syncthreads();  // all reads of aState/aOut complete

        // GRU gates + state update (wave-local columns), write tar + next A-frags
        for (int v = 0; v < 8; ++v) {
            float z = sigmoidf_fast(accZ[v]);
            float r = sigmoidf_fast(accR[v]);
            float c = tanhf_fast(accXH[v] + r * accHH[v]);
            float ns = z * stC[v] + (1.0f - z) * c;
            stC[v] = ns;
            int row = v + ((lane >= 16) ? 8 : 0);
            tar[((size_t)(b0 + row) * TARP + (Pn + t)) * Hn + col] = ns;
            aState[wkt][row + wl][ws_] = f2b(ns);
        }
        __syncthreads();  // new state fragments visible

        // new_out = concat(new_state, attn) @ Wlin — wave 0 only, K=1024.
        // 4 independent accumulator chains (state/attn x low/high K) to break the
        // 32-deep WMMA RAW dependency chain while 31 waves wait at the barrier.
        if (wave == 0) {
            const u16* wl0 = wlFrag + zz + lane * 16;
            v8f acc0, acc1, acc2, acc3;
            for (int v = 0; v < 8; ++v) { acc0[v] = bO; acc1[v] = 0.f; acc2[v] = 0.f; acc3[v] = 0.f; }
            for (int kt = 0; kt < 8; ++kt) {
                v16bf aS0 = *(const v16bf*)&aState[kt][lane][0];
                v16bf aS1 = *(const v16bf*)&aState[kt + 8][lane][0];
                v16bf aA0 = *(const v16bf*)&aAttn[kt][lane][0];
                v16bf aA1 = *(const v16bf*)&aAttn[kt + 8][lane][0];
                acc0 = wmma_bf16(aS0, *(const v16bf*)(wl0 + (size_t)kt * 512),        acc0);
                acc1 = wmma_bf16(aS1, *(const v16bf*)(wl0 + (size_t)(kt + 8) * 512),  acc1);
                acc2 = wmma_bf16(aA0, *(const v16bf*)(wl0 + (size_t)(kt + 16) * 512), acc2);
                acc3 = wmma_bf16(aA1, *(const v16bf*)(wl0 + (size_t)(kt + 24) * 512), acc3);
            }
            v8f accO;
            for (int v = 0; v < 8; ++v) accO[v] = (acc0[v] + acc1[v]) + (acc2[v] + acc3[v]);
            int c = lane & 15;
            if (c < Dn) {
                int hh2 = (c >> 3) & 1, ss2 = c & 7;
                for (int v = 0; v < 8; ++v) {
                    int row = v + ((lane >= 16) ? 8 : 0);
                    float o = accO[v];
                    seq[((size_t)(b0 + row) * Sn + (t + 1)) * Dn + c] = o;
                    aOut[row + 16 * hh2][ss2] = f2b(o);   // K>=15 slots stay zero from prologue
                }
            }
        }
        __syncthreads();  // aOut ready for next step
    }
}

extern "C" void kernel_launch(void* const* d_in, const int* in_sizes, int n_in,
                              void* d_out, int out_size, void* d_ws, size_t ws_size,
                              hipStream_t stream) {
    const float* hs        = (const float*)d_in[0];
    const float* masks     = (const float*)d_in[1];
    const float* init_pt   = (const float*)d_in[2];
    const float* Wq        = (const float*)d_in[3];
    const float* bq        = (const float*)d_in[4];
    const float* Wemb      = (const float*)d_in[5];
    const float* bemb      = (const float*)d_in[6];
    const float* gru_k     = (const float*)d_in[7];
    const float* gru_rk    = (const float*)d_in[8];
    const float* gru_b     = (const float*)d_in[9];
    const float* Wlin      = (const float*)d_in[10];
    const float* blin      = (const float*)d_in[11];

    float* seq = (float*)d_out;
    float* tar = seq + (size_t)Bsz * Sn * Dn;

    char* ws = (char*)d_ws;
    float* attn    = (float*)(ws + 0);        // 262144 B
    float* wfStage = (float*)(ws + 262144);   // 92160 B (15x1536 f32)
    float* bias_x  = (float*)(ws + 354304);   // 6144 B
    u16*   wfFrag  = (u16*)  (ws + 360448);   // 98304 B  (96 frags)
    u16*   rkFrag  = (u16*)  (ws + 458752);   // 1572864 B (96x16 frags)
    u16*   wlFrag  = (u16*)  (ws + 2031616);  // 32768 B  (32 frags)

    // setup: weight fusion + bf16 fragment packing (all L2-resident afterwards)
    k_wfused<<<(Dn * TH3 + 255) / 256, 256, 0, stream>>>(Wemb, gru_k, wfStage);
    k_biasx<<<(TH3 + 255) / 256, 256, 0, stream>>>(bemb, gru_k, gru_b, bias_x);
    k_pack_b<<<(96 * 1 * 512 + 255) / 256, 256, 0, stream>>>(wfStage, Dn, TH3, TH3, 1, 96, wfFrag);
    k_pack_b<<<(96 * 16 * 512 + 255) / 256, 256, 0, stream>>>(gru_rk, Hn, TH3, TH3, 16, 96, rkFrag);
    k_pack_b<<<(1 * 32 * 512 + 255) / 256, 256, 0, stream>>>(Wlin, 2 * Hn, Dn, Dn, 32, 1, wlFrag);

    // attention + bulk copy of hidden_states into tar
    k_attention<<<Bsz, 256, 0, stream>>>(hs, masks, Wq, bq, attn);
    {
        size_t total4 = (size_t)Bsz * Pn * Hn / 4;
        k_copy_hs<<<(unsigned)((total4 + 255) / 256), 256, 0, stream>>>((const float4*)hs, (float4*)tar);
    }

    // persistent sequential scan: 8 WGs (16 batch rows each), 32 waves per WG
    k_scan<<<Bsz / 16, 1024, 0, stream>>>(hs, init_pt, attn, wfFrag, rkFrag, wlFrag,
                                          bias_x, gru_b, blin, seq, tar);
}
